// MultiHeadAttention_54382875902378
// MI455X (gfx1250) — matrix operand order repaired and checked by reading, not compiled, not run on hardware
//
#include <hip/hip_runtime.h>
#include <hip/hip_bf16.h>
#include <math.h>

// ---------------------------------------------------------------------------
// MI455X (gfx1250) multi-head attention, bf16 WMMA + async-to-LDS pipeline.
//   stage 0: fp32 -> bf16 conversion (weights transposed to [N][K])
//   stage 1: Q/K/V projections (WMMA, async double-buffered LDS staging);
//            Q projection folds the 1/sqrt(Dh) softmax scale into its store;
//            V projection stores head-transposed Vt[d][t]
//   stage 2: flash attention per (b,h): L=2048, Dh=64, online softmax, WMMA;
//            row-sum computed by a ones-vector WMMA (5th accumulator tile)
//   stage 3: output projection -> fp32 d_out (WMMA)
// ---------------------------------------------------------------------------

typedef __attribute__((ext_vector_type(8)))  float          v8f;
typedef __attribute__((ext_vector_type(16))) __bf16         v16bf;
typedef __attribute__((ext_vector_type(16))) unsigned short u16x16;
typedef __attribute__((ext_vector_type(4)))  unsigned int   u32x4;

static __device__ __forceinline__ unsigned short f2bf(float f) {
  unsigned u = __builtin_bit_cast(unsigned, f);
  u += 0x7FFFu + ((u >> 16) & 1u);            // round-to-nearest-even
  return (unsigned short)(u >> 16);
}

// Two contiguous 16-byte runs -> one 16-element bf16 fragment.
static __device__ __forceinline__ v16bf ld_frag(const unsigned short* p0,
                                                const unsigned short* p1) {
  union { u32x4 q[2]; v16bf f; } u;
  u.q[0] = *(const u32x4*)p0;
  u.q[1] = *(const u32x4*)p1;
  return u.f;
}

static __device__ __forceinline__ unsigned lds_off(const void* p) {
  // flat LDS aperture: low 32 bits == workgroup-relative LDS byte offset
  return (unsigned)(unsigned long long)p;
}
static __device__ __forceinline__ unsigned long long gaddr(const void* p) {
  return (unsigned long long)p;
}
// ASYNCcnt-tracked 16B global -> LDS copy (per-lane addresses).
static __device__ __forceinline__ void async_b128(unsigned lds, unsigned long long g) {
  asm volatile("global_load_async_to_lds_b128 %0, %1, off" :: "v"(lds), "v"(g) : "memory");
}
static __device__ __forceinline__ void wait_async(int pending) {
  if (pending) asm volatile("s_wait_asynccnt 0x4" ::: "memory");
  else         asm volatile("s_wait_asynccnt 0x0" ::: "memory");
}
// Keep the fragment-load group and the WMMA group from being interleaved.
static __device__ __forceinline__ void sched_fence() {
#if __has_builtin(__builtin_amdgcn_sched_barrier)
  __builtin_amdgcn_sched_barrier(0);
#endif
}

__global__ void cvt_f32_to_bf16(const float* __restrict__ in,
                                unsigned short* __restrict__ out, int n) {
  int i = blockIdx.x * blockDim.x + threadIdx.x;
  if (i < n) out[i] = f2bf(in[i]);
}

// in: K x N row-major fp32  ->  out: N x K row-major bf16
__global__ void cvt_transpose_f32_to_bf16(const float* __restrict__ in,
                                          unsigned short* __restrict__ out,
                                          int K, int N) {
  int idx = blockIdx.x * blockDim.x + threadIdx.x;
  if (idx < K * N) {
    int k = idx / N, n = idx - k * N;
    out[(size_t)n * K + k] = f2bf(in[idx]);
  }
}

// ---------------------------------------------------------------------------
// C(MxN) = A(MxK) * Bt(NxK)^T + bias(N).  Block 256 thr = 8 waves,
// tile 128x128, K-step 32, double-buffered async LDS staging.
// MODE 0: bf16 row-major; 1: fp32 row-major;
// MODE 2: bf16 head-transposed Vt; 3: bf16 row-major scaled by 1/8 (Q).
// ---------------------------------------------------------------------------
template <int MODE>
__global__ __launch_bounds__(256)
void gemm_bf16_wmma(const unsigned short* __restrict__ A, int lda,
                    const unsigned short* __restrict__ Bt,   // N x K
                    const float* __restrict__ bias,
                    void* __restrict__ Cout, int ldc, int Kdim) {
  __shared__ alignas(16) unsigned short As[2][128 * 32];  // [m][k]
  __shared__ alignas(16) unsigned short Bs[2][128 * 32];  // [n][k]

  const int tid  = threadIdx.x;
  const int lane = tid & 31;
  const int wave = tid >> 5;
  const int half = lane >> 4;
  const int r    = lane & 15;

  const int rowBase = blockIdx.y * 128;
  const int colBase = blockIdx.x * 128;
  const int wm = (wave >> 2) * 64;
  const int wn = (wave & 3) * 32;

  auto stage = [&](int buf, int k0) {
#pragma unroll
    for (int u = 0; u < 2; ++u) {
      const int idx = tid * 2 + u;          // 0..511
      const int row = idx >> 2, seg = idx & 3;
      async_b128(lds_off(&As[buf][row * 32 + seg * 8]),
                 gaddr(A + (size_t)(rowBase + row) * lda + k0 + seg * 8));
      async_b128(lds_off(&Bs[buf][row * 32 + seg * 8]),
                 gaddr(Bt + (size_t)(colBase + row) * Kdim + k0 + seg * 8));
    }
  };

  v8f acc[4][2] = {};
  stage(0, 0);
  for (int k0 = 0; k0 < Kdim; k0 += 32) {
    const int cur = (k0 >> 5) & 1;
    const int more = (k0 + 32 < Kdim);
    if (more) stage(cur ^ 1, k0 + 32);
    wait_async(more);
    __syncthreads();

    v16bf afr[4], bfr[2];
#pragma unroll
    for (int mi = 0; mi < 4; ++mi) {        // A frag: lane = M row
      const unsigned short* p = &As[cur][(wm + mi * 16 + r) * 32];
      afr[mi] = ld_frag(p + 8 * half, p + 16 + 8 * half);
    }
#pragma unroll
    for (int ni = 0; ni < 2; ++ni) {        // B frag: lane = N col
      const unsigned short* p = &Bs[cur][(wn + ni * 16 + r) * 32];
      bfr[ni] = ld_frag(p + 8 * half, p + 16 + 8 * half);
    }
    sched_fence();                          // batch DS loads under one wait
#pragma unroll
    for (int mi = 0; mi < 4; ++mi)
#pragma unroll
      for (int ni = 0; ni < 2; ++ni)
        acc[mi][ni] = __builtin_amdgcn_wmma_f32_16x16x32_bf16(
            false, afr[mi], false, bfr[ni], (short)0, acc[mi][ni], false, false);
    sched_fence();
    __syncthreads();
  }

  constexpr float oscale = (MODE == 3) ? 0.125f : 1.0f;   // fold 1/sqrt(Dh)
#pragma unroll
  for (int mi = 0; mi < 4; ++mi) {
#pragma unroll
    for (int ni = 0; ni < 2; ++ni) {
      const int col = colBase + wn + ni * 16 + r;
      const float bia = bias[col];
#pragma unroll
      for (int i = 0; i < 8; ++i) {         // C slot i -> row i + 8*half
        const int row = rowBase + wm + mi * 16 + i + 8 * half;
        const float v = (acc[mi][ni][i] + bia) * oscale;
        if constexpr (MODE == 1) {
          ((float*)Cout)[(size_t)row * ldc + col] = v;
        } else if constexpr (MODE == 2) {
          // row=(b,s), col=c -> head-transposed Vt[(b,h)][d][t], t stride 1
          const int b = row >> 8, s = row & 255;
          const int h = (col >> 6) & 7, d = col & 63, t = s * 8 + (col >> 9);
          ((unsigned short*)Cout)[(size_t)(b * 8 + h) * 131072 +
                                  (size_t)d * 2048 + t] = f2bf(v);
        } else {
          ((unsigned short*)Cout)[(size_t)row * ldc + col] = f2bf(v);
        }
      }
    }
  }
}

// ---------------------------------------------------------------------------
// Flash attention. Q/K head-strided: elem (t,d) at b*2^20 + h*64 + t*512 + d.
// Vt head-transposed: elem (d,t) at (b*8+h)*131072 + d*2048 + t.
// Block 128 thr = 4 waves; wave owns 16 q rows; kv tiles of 32, async-staged.
// Q is pre-scaled by 1/8, so scores come out of WMMA already scaled.
// ---------------------------------------------------------------------------
__global__ __launch_bounds__(128)
void attn_flash_wmma(const unsigned short* __restrict__ Qp,
                     const unsigned short* __restrict__ Kp,
                     const unsigned short* __restrict__ Vt,
                     unsigned short* __restrict__ Ctx) {
  __shared__ alignas(16) unsigned short Ks[2][32 * 64];   // [kv][d]
  __shared__ alignas(16) unsigned short Vs[2][64 * 32];   // [d][kv]
  __shared__ alignas(16) unsigned short Plds[4][16 * 32]; // per-wave probs

  const int tid  = threadIdx.x;
  const int lane = tid & 31, wave = tid >> 5;
  const int half = lane >> 4, r = lane & 15;

  const int L = 2048, RS = 512;
  const int qt = blockIdx.x & 31;
  const int bh = blockIdx.x >> 5;
  const int b = bh >> 3, h = bh & 7;
  const size_t baseQ = (size_t)b * 1048576 + (size_t)h * 64;
  const size_t baseV = (size_t)bh * 131072;
  const int t0 = qt * 64 + wave * 16;

  auto stage = [&](int buf, int kv) {
#pragma unroll
    for (int u = 0; u < 2; ++u) {
      const int idx = tid * 2 + u;          // 0..255
      const int kr = idx >> 3, ksg = idx & 7;   // K: 32 rows x 8 x 16B
      async_b128(lds_off(&Ks[buf][kr * 64 + ksg * 8]),
                 gaddr(Kp + baseQ + (size_t)(kv + kr) * RS + ksg * 8));
      const int vr = idx >> 2, vsg = idx & 3;   // Vt: 64 rows x 4 x 16B
      async_b128(lds_off(&Vs[buf][vr * 32 + vsg * 8]),
                 gaddr(Vt + baseV + (size_t)vr * 2048 + kv + vsg * 8));
    }
  };

  // Q fragments: 16x64 as two 16x32 A-frags (contiguous runs)
  v16bf qf[2];
#pragma unroll
  for (int c = 0; c < 2; ++c) {
    const unsigned short* qrow = Qp + baseQ + (size_t)(t0 + r) * RS + c * 32;
    qf[c] = ld_frag(qrow + 8 * half, qrow + 16 + 8 * half);
  }

  // all-ones B fragment: row-sum of P via WMMA
  v16bf ones;
  {
    u16x16 u;
#pragma unroll
    for (int i = 0; i < 16; ++i) u[i] = 0x3F80;   // bf16 1.0
    ones = __builtin_bit_cast(v16bf, u);
  }

  v8f acc[4] = {};
  v8f accl = {};                       // running row-sum tile
  float mrow[8];
#pragma unroll
  for (int i = 0; i < 8; ++i) mrow[i] = -INFINITY;

  stage(0, 0);
  for (int kv = 0; kv < L; kv += 32) {
    const int cur = (kv >> 5) & 1;
    const int more = (kv + 32 < L);
    if (more) stage(cur ^ 1, kv + 32);
    wait_async(more);
    __syncthreads();

    // preload all K fragments, then chain WMMAs (loads batch under one wait)
    v16bf kf[2][2];
#pragma unroll
    for (int nt = 0; nt < 2; ++nt)
#pragma unroll
      for (int c = 0; c < 2; ++c) {
        const unsigned short* p =
            &Ks[cur][(nt * 16 + r) * 64 + c * 32];
        kf[nt][c] = ld_frag(p + 8 * half, p + 16 + 8 * half);
      }
    sched_fence();
    v8f s[2] = {};
#pragma unroll
    for (int nt = 0; nt < 2; ++nt)
#pragma unroll
      for (int c = 0; c < 2; ++c)
        s[nt] = __builtin_amdgcn_wmma_f32_16x16x32_bf16(
            false, qf[c], false, kf[nt][c], (short)0, s[nt], false, false);
    sched_fence();

    // online softmax; row slot i lives in one 16-lane half -> xor 1,2,4,8
#pragma unroll
    for (int i = 0; i < 8; ++i) {
      float mx = fmaxf(s[0][i], s[1][i]);
#pragma unroll
      for (int off = 1; off < 16; off <<= 1)
        mx = fmaxf(mx, __shfl_xor(mx, off, 32));
      const float mnew  = fmaxf(mrow[i], mx);
      const float alpha = __expf(mrow[i] - mnew);
      mrow[i] = mnew;
      const float p0 = __expf(s[0][i] - mnew);
      const float p1 = __expf(s[1][i] - mnew);
#pragma unroll
      for (int t = 0; t < 4; ++t) acc[t][i] *= alpha;
      accl[i] *= alpha;
      const int prow = i + 8 * half;
      Plds[wave][prow * 32 + r]      = f2bf(p0);
      Plds[wave][prow * 32 + 16 + r] = f2bf(p1);
    }
    asm volatile("s_wait_dscnt 0" ::: "memory");

    const unsigned short* pp = &Plds[wave][r * 32];
    v16bf pf = ld_frag(pp + 8 * half, pp + 16 + 8 * half);  // P as A-frag
    v16bf vf[4];
#pragma unroll
    for (int nt = 0; nt < 4; ++nt) {
      const unsigned short* p = &Vs[cur][(nt * 16 + r) * 32];
      vf[nt] = ld_frag(p + 8 * half, p + 16 + 8 * half);
    }
    sched_fence();
#pragma unroll
    for (int nt = 0; nt < 4; ++nt)           // ctx += P * V  (Vt rows = d)
      acc[nt] = __builtin_amdgcn_wmma_f32_16x16x32_bf16(
          false, pf, false, vf[nt], (short)0, acc[nt], false, false);
    accl = __builtin_amdgcn_wmma_f32_16x16x32_bf16(   // l += rowsum(P)
        false, pf, false, ones, (short)0, accl, false, false);
    sched_fence();
    __syncthreads();
  }

#pragma unroll
  for (int i = 0; i < 8; ++i) {
    const float inv = 1.0f / accl[i];        // every col holds the row-sum
    const int trow = t0 + i + 8 * half;
#pragma unroll
    for (int nt = 0; nt < 4; ++nt)
      Ctx[baseQ + (size_t)trow * RS + nt * 16 + r] = f2bf(acc[nt][i] * inv);
  }
}

// ---------------------------------------------------------------------------
extern "C" void kernel_launch(void* const* d_in, const int* in_sizes, int n_in,
                              void* d_out, int out_size, void* d_ws, size_t ws_size,
                              hipStream_t stream) {
  (void)in_sizes; (void)n_in; (void)out_size; (void)ws_size;
  const float* Q  = (const float*)d_in[0];
  const float* K  = (const float*)d_in[1];
  const float* V  = (const float*)d_in[2];
  const float* Wq = (const float*)d_in[3];
  const float* bq = (const float*)d_in[4];
  const float* Wk = (const float*)d_in[5];
  const float* bk = (const float*)d_in[6];
  const float* Wv = (const float*)d_in[7];
  const float* bv = (const float*)d_in[8];
  const float* Wf = (const float*)d_in[9];
  const float* bf = (const float*)d_in[10];

  const int BS = 512, D = 512, N4 = 4096;
  const size_t szW = (size_t)D * N4;   // 2,097,152
  const size_t szX = (size_t)BS * D;   // 262,144
  const size_t szP = (size_t)BS * N4;  // 2,097,152

  unsigned short* w    = (unsigned short*)d_ws;
  unsigned short* Wqt  = w; w += szW;   // 4096 x 512
  unsigned short* Wkt  = w; w += szW;
  unsigned short* Wvt  = w; w += szW;
  unsigned short* Wft  = w; w += szW;   // 512 x 4096
  unsigned short* Qb   = w; w += szX;
  unsigned short* Kb   = w; w += szX;
  unsigned short* Vb   = w; w += szX;
  unsigned short* Qp   = w; w += szP;   // strided head layout, pre-scaled 1/8
  unsigned short* Kp   = w; w += szP;
  unsigned short* Vpt  = w; w += szP;   // head-transposed [bh][d][t]
  unsigned short* Ctx  = w; w += szP;

  auto cvt = [&](const float* in, unsigned short* out, size_t n) {
    cvt_f32_to_bf16<<<dim3((unsigned)((n + 255) / 256)), dim3(256), 0, stream>>>(
        in, out, (int)n);
  };
  auto cvtT = [&](const float* in, unsigned short* out, int Kd, int Nd) {
    size_t n = (size_t)Kd * Nd;
    cvt_transpose_f32_to_bf16<<<dim3((unsigned)((n + 255) / 256)), dim3(256), 0,
                                stream>>>(in, out, Kd, Nd);
  };
  cvtT(Wq, Wqt, D, N4); cvtT(Wk, Wkt, D, N4); cvtT(Wv, Wvt, D, N4);
  cvtT(Wf, Wft, N4, D);
  cvt(Q, Qb, szX); cvt(K, Kb, szX); cvt(V, Vb, szX);

  // projections: (512x512)@(512x4096)+b
  dim3 gproj(N4 / 128, BS / 128);
  gemm_bf16_wmma<3><<<gproj, 256, 0, stream>>>(Qb, D, Wqt, bq, Qp, N4, D);
  gemm_bf16_wmma<0><<<gproj, 256, 0, stream>>>(Kb, D, Wkt, bk, Kp, N4, D);
  gemm_bf16_wmma<2><<<gproj, 256, 0, stream>>>(Vb, D, Wvt, bv, Vpt, N4, D);

  // flash attention: 16 (b,h) * 32 q-tiles
  attn_flash_wmma<<<dim3(512), dim3(128), 0, stream>>>(Qp, Kp, Vpt, Ctx);

  // output projection: (512x4096)@(4096x512)+bf -> fp32 d_out
  dim3 gout(D / 128, BS / 128);
  gemm_bf16_wmma<1><<<gout, 256, 0, stream>>>(Ctx, N4, Wft, bf, d_out, D, N4);
}
